// BIKVAttention_27066883899897
// MI455X (gfx1250) — matrix operand order repaired and checked by reading, not compiled, not run on hardware
//
#include <hip/hip_runtime.h>
#include <hip/hip_bf16.h>

typedef __bf16 bf16;
typedef __attribute__((ext_vector_type(16))) __bf16 v16bf;
typedef __attribute__((ext_vector_type(8)))  __bf16 v8bf;
typedef __attribute__((ext_vector_type(8)))  float  v8f;
typedef __attribute__((ext_vector_type(4)))  unsigned int u32x4;
typedef __attribute__((ext_vector_type(8)))  int i32x8;
typedef __attribute__((ext_vector_type(4)))  int i32x4;

#define DEVINL __device__ __forceinline__

// ---------------------------------------------------------------------------
// WMMA helpers (CDNA5 layouts, CDNA5 ISA §7.12.2)
// ---------------------------------------------------------------------------
DEVINL v8f wmma_bf16(v16bf a, v16bf b, v8f c) {
  return __builtin_amdgcn_wmma_f32_16x16x32_bf16(false, a, false, b, (short)0, c,
                                                 false, false);
}

// A matrix 16x32 bf16: lane m(0-15) holds row m, K {0..15};
// lane m+16 holds row m, K {16..31}.  Two 16B loads.
DEVINL v16bf load_a_frag(const bf16* base, int ldk, int lane) {
  int m = lane & 15, h = lane >> 4;
  const bf16* p = base + (size_t)m * ldk + h * 16;
  union { v16bf v; v8bf q[2]; } u;
  u.q[0] = *(const v8bf*)(p);
  u.q[1] = *(const v8bf*)(p + 8);
  return u.v;
}

// B matrix 32x16 bf16 with W stored [N,K]: lane l holds column n=l%16,
// K-half = l/16 -> one contiguous 32B read of W row n.
DEVINL v16bf load_b_frag(const bf16* base, int ldk, int lane) {
  int n = lane & 15, h = lane >> 4;
  return *(const v16bf*)(base + (size_t)n * ldk + h * 16);
}

// ---------------------------------------------------------------------------
// fp32 -> bf16 convert (grid-stride)
// ---------------------------------------------------------------------------
__global__ void cvt_bf16_kernel(const float* __restrict__ in, bf16* __restrict__ o,
                                size_t n) {
  size_t t = (size_t)blockIdx.x * blockDim.x + threadIdx.x;
  size_t stride = (size_t)gridDim.x * blockDim.x;
  for (; t < n; t += stride) o[t] = (bf16)in[t];
}

// ---------------------------------------------------------------------------
// Generic WMMA GEMM: D[M,N] = act(A[M,K] @ W[N,K]^T) (+ biasVec[n])
// one 16x16 tile per wave, 8 waves per block
// ---------------------------------------------------------------------------
__global__ void gemm_bf16_kernel(const bf16* __restrict__ A, const bf16* __restrict__ W,
                                 float* __restrict__ Df, bf16* __restrict__ Db,
                                 const float* __restrict__ biasVec,
                                 int M, int N, int K, int sigmoid, int transD) {
  int lane = threadIdx.x & 31;
  int wave = threadIdx.x >> 5;
  int tilesN = N >> 4;
  int tile = blockIdx.x * 8 + wave;
  if (tile >= (M >> 4) * tilesN) return;
  int m16 = tile / tilesN, n16 = tile - m16 * tilesN;
  const bf16* Ab = A + (size_t)m16 * 16 * K;
  const bf16* Wb = W + (size_t)n16 * 16 * K;
  v8f acc = {0.f, 0.f, 0.f, 0.f, 0.f, 0.f, 0.f, 0.f};
  for (int k0 = 0; k0 < K; k0 += 32) {
    v16bf a = load_a_frag(Ab + k0, K, lane);
    v16bf b = load_b_frag(Wb + k0, K, lane);
    acc = wmma_bf16(a, b, acc);
  }
  int nl = lane & 15, hf = lane >> 4;
  int col = n16 * 16 + nl;
  float bv = biasVec ? biasVec[col] : 0.0f;
#pragma unroll
  for (int v = 0; v < 8; ++v) {
    int row = m16 * 16 + v + 8 * hf;
    float val = acc[v] + bv;
    if (sigmoid) val = 1.0f / (1.0f + __expf(-val));
    if (Df) Df[(size_t)row * N + col] = val;
    if (Db) {
      if (transD) Db[(size_t)col * M + row] = (bf16)val;
      else        Db[(size_t)row * N + col] = (bf16)val;
    }
  }
}

// ---------------------------------------------------------------------------
// sim = idx @ indices_tab^T fused with argmax, TDM-staged codebook panels.
// Grid = 16 query-groups x 8 codebook slices. Each block: 8 waves own 8
// distinct 16-row query tiles; a 16-entry x 512 panel (16KB) is DMA'd into
// LDS by the Tensor Data Mover (double-buffered) and shared by all waves
// -> 128 WMMAs per panel instead of 16.
// ---------------------------------------------------------------------------
__global__ void sim_argmax_tdm_kernel(const bf16* __restrict__ idxbf,
                                      const bf16* __restrict__ tabbf,
                                      float* __restrict__ partVal,
                                      int* __restrict__ partIdx) {
  const int KD = 512;
  const int SLICE = 8192;           // codebook entries per block
  const int PANEL = 16;             // entries per LDS panel
  const int NIT = SLICE / PANEL;    // 512 iterations
  int lane = threadIdx.x & 31;
  int wave = threadIdx.x >> 5;
  int qgroup = blockIdx.x >> 3;     // 0..15
  int slice  = blockIdx.x & 7;      // 0..7
  int qtile = qgroup * 8 + wave;    // 0..127

  __shared__ __align__(1024) bf16 panel[2][PANEL * KD];   // 2 x 16KB

  // preload the wave's private A fragments (16 rows x 512) in registers
  const bf16* Ab = idxbf + (size_t)qtile * 16 * KD;
  v16bf afrag[16];
#pragma unroll
  for (int kk = 0; kk < 16; ++kk) afrag[kk] = load_a_frag(Ab + kk * 32, KD, lane);

  unsigned long long tabBase = (unsigned long long)(size_t)tabbf;
  unsigned lds0 = (unsigned)(size_t)&panel[0][0];
  unsigned lds1 = (unsigned)(size_t)&panel[1][0];

  // Tensor DMA descriptor per CDNA5 ISA:
  // 2-D tensor [65536 x 512], data_size=2B, tile = 16 rows x 512 cols.
  auto tdm_issue = [&](int it) {
    int n0 = slice * SLICE + it * PANEL;
    unsigned long long g = tabBase + (unsigned long long)n0 * (KD * 2);
    unsigned lds = (it & 1) ? lds1 : lds0;
    u32x4 g0;
    g0[0] = 1u;                                           // count=1
    g0[1] = lds;                                          // lds_addr
    g0[2] = (unsigned)(g & 0xFFFFFFFFu);                  // global_addr[31:0]
    g0[3] = (unsigned)((g >> 32) & 0x1FFFFFFu) | (2u << 30); // addr[56:32]|type=2
    i32x8 g1;
    g1[0] = (1 << 16);                                    // data_size=1 (2 bytes)
    g1[1] = (KD << 16);                                   // tensor_dim0=512
    g1[2] = 0;                                            // dim0 hi / dim1 lo16(65536)=0
    g1[3] = 1 | (KD << 16);                               // dim1 hi16=1, tile_dim0=512
    g1[4] = PANEL;                                        // tile_dim1=16, tile_dim2=0
    g1[5] = KD;                                           // tensor_dim0_stride=512
    g1[6] = 0;
    g1[7] = 0;
    i32x4 g2 = {0, 0, 0, 0};
    i32x4 g3 = {0, 0, 0, 0};
    i32x8 g4 = {0, 0, 0, 0, 0, 0, 0, 0};                  // extra group (clang-23 form)
    __builtin_amdgcn_tensor_load_to_lds(g0, g1, g2, g3, g4, 0);
  };

  if (wave == 0) {
    tdm_issue(0);
    __builtin_amdgcn_s_wait_tensorcnt(0);
  }
  __syncthreads();

  float mx[8]; int mi[8];
#pragma unroll
  for (int v = 0; v < 8; ++v) { mx[v] = -3.0e38f; mi[v] = 0; }

  for (int it = 0; it < NIT; ++it) {
    if (wave == 0 && it + 1 < NIT) tdm_issue(it + 1);     // overlap next DMA
    const bf16* P = &panel[it & 1][0];
    v8f acc = {0.f, 0.f, 0.f, 0.f, 0.f, 0.f, 0.f, 0.f};
    int n = lane & 15, hfk = lane >> 4;
#pragma unroll
    for (int kk = 0; kk < 16; ++kk) {
      const bf16* p = P + (size_t)n * KD + kk * 32 + hfk * 16;
      union { v16bf v; v8bf q[2]; } u;
      u.q[0] = *(const v8bf*)(p);
      u.q[1] = *(const v8bf*)(p + 8);
      acc = wmma_bf16(afrag[kk], u.v, acc);
    }
    int e = slice * SLICE + it * PANEL + n;
#pragma unroll
    for (int v = 0; v < 8; ++v) {
      float val = acc[v];
      if (val > mx[v]) { mx[v] = val; mi[v] = e; }
    }
    if (wave == 0) __builtin_amdgcn_s_wait_tensorcnt(0);
    __syncthreads();                                      // panel ready / reusable
  }

  // argmax across the 16 lanes that hold one row (xor masks < 16 stay in-group)
#pragma unroll
  for (int v = 0; v < 8; ++v) {
#pragma unroll
    for (int off = 1; off < 16; off <<= 1) {
      float ov = __shfl_xor(mx[v], off, 32);
      int   oi = __shfl_xor(mi[v], off, 32);
      if (ov > mx[v] || (ov == mx[v] && oi < mi[v])) { mx[v] = ov; mi[v] = oi; }
    }
  }
  int hf = lane >> 4;
  if ((lane & 15) == 0) {
#pragma unroll
    for (int v = 0; v < 8; ++v) {
      int row = qtile * 16 + v + 8 * hf;
      partVal[(size_t)row * 8 + slice] = mx[v];
      partIdx[(size_t)row * 8 + slice] = mi[v];
    }
  }
}

// ---------------------------------------------------------------------------
// Reduce the 8 per-slice argmax partials per row (first-max tie-break)
// ---------------------------------------------------------------------------
__global__ void argmax_reduce_kernel(const float* __restrict__ partVal,
                                     const int* __restrict__ partIdx,
                                     int* __restrict__ choices, int rows) {
  int r = blockIdx.x * blockDim.x + threadIdx.x;
  if (r >= rows) return;
  float best = partVal[(size_t)r * 8];
  int bi = partIdx[(size_t)r * 8];
  for (int s2 = 1; s2 < 8; ++s2) {
    float ov = partVal[(size_t)r * 8 + s2];
    int oi = partIdx[(size_t)r * 8 + s2];
    if (ov > best || (ov == best && oi < bi)) { best = ov; bi = oi; }
  }
  choices[r] = bi;
}

// ---------------------------------------------------------------------------
// Gather key/value/index-table rows by choices (fp32 -> bf16 on the fly)
// ---------------------------------------------------------------------------
__global__ void gather_kernel(const int* __restrict__ choices,
                              const float* __restrict__ key_tab,
                              const float* __restrict__ val_tab,
                              const bf16* __restrict__ tabbf,
                              bf16* __restrict__ ckbf, bf16* __restrict__ cvbf,
                              bf16* __restrict__ gidxbf) {
  int t = blockIdx.x * blockDim.x + threadIdx.x;  // exactly 2048*512 threads
  int row = t >> 9, col = t & 511;
  int e = choices[row];
  size_t src = (size_t)e * 512 + col;
  ckbf[t]   = (bf16)key_tab[src];
  cvbf[t]   = (bf16)val_tab[src];
  gidxbf[t] = tabbf[src];
}

// ---------------------------------------------------------------------------
// Rotary (tables indexed by HEAD per the reference quirk) + pack to bf16
// ---------------------------------------------------------------------------
__global__ void rope_kernel(const float* __restrict__ x, bf16* __restrict__ o,
                            int total) {
  int t = blockIdx.x * blockDim.x + threadIdx.x;
  if (t >= total) return;
  int col = t & 511;       // H = 512
  int hd = col & 63;       // dim within head
  int hh = col >> 6;       // head index (used as "position")
  int dd = hd & 31;
  float invf = __powf(10000.0f, -(float)(2 * dd) / 64.0f);
  float ang = (float)hh * invf;
  float s, c;
  __sincosf(ang, &s, &c);
  int base = t - col;
  float xv = x[t];
  float pr = (hd < 32) ? -x[base + col + 32] : x[base + col - 32];
  o[t] = (bf16)(xv * c + pr * s);
}

// ---------------------------------------------------------------------------
// Flash-style attention with learned bias + causal mask.
// One wave per (b, h, 16-query block); 32 keys per iteration; P goes through
// a private LDS slice for the D-layout -> A-layout transpose (in-wave LDS
// ordering, no barrier needed).
// ---------------------------------------------------------------------------
__global__ void attention_kernel(const bf16* __restrict__ qbf,
                                 const bf16* __restrict__ kbf,
                                 const bf16* __restrict__ vT,  // [H, B*S]
                                 const float* __restrict__ bias,
                                 bf16* __restrict__ obf) {
  const int S = 1024, H = 512, HD = 64;
  int lane = threadIdx.x & 31;
  int wave = threadIdx.x >> 5;
  int wg = blockIdx.x * 8 + wave;   // 0..1023
  int qblk = wg & 63;
  int h = (wg >> 6) & 7;
  int b = wg >> 9;
  int qBase = qblk << 4;
  int nl = lane & 15, hf = lane >> 4;
  __shared__ __align__(64) bf16 pshare[8][16][32];

  const bf16* Qb = qbf + ((size_t)(b * S + qBase)) * H + h * HD;
  v16bf aq0 = load_a_frag(Qb, H, lane);
  v16bf aq1 = load_a_frag(Qb + 32, H, lane);

  v8f oacc[4] = {};
  float mrow[8], lrow[8];
#pragma unroll
  for (int v = 0; v < 8; ++v) { mrow[v] = -3.0e38f; lrow[v] = 0.f; }

  for (int kb = 0; kb < qBase + 16; kb += 32) {
    v8f s0 = {0.f, 0.f, 0.f, 0.f, 0.f, 0.f, 0.f, 0.f};
    v8f s1 = {0.f, 0.f, 0.f, 0.f, 0.f, 0.f, 0.f, 0.f};
    const bf16* Kb0 = kbf + ((size_t)(b * S + kb)) * H + h * HD;
    {
      v16bf bk0 = load_b_frag(Kb0, H, lane);
      v16bf bk1 = load_b_frag(Kb0 + (size_t)16 * H, H, lane);
      s0 = wmma_bf16(aq0, bk0, s0);
      s1 = wmma_bf16(aq0, bk1, s1);
      bk0 = load_b_frag(Kb0 + 32, H, lane);
      bk1 = load_b_frag(Kb0 + (size_t)16 * H + 32, H, lane);
      s0 = wmma_bf16(aq1, bk0, s0);
      s1 = wmma_bf16(aq1, bk1, s1);
    }
    int key0 = kb + nl, key1 = key0 + 16;
#pragma unroll
    for (int v = 0; v < 8; ++v) {
      int q = qBase + v + 8 * hf;
      const float* brow = bias + ((size_t)b * S + q) * S;
      float t0 = s0[v] * 0.125f + brow[key0];
      float t1 = s1[v] * 0.125f + brow[key1];
      if (key0 > q) t0 = -3.0e38f;
      if (key1 > q) t1 = -3.0e38f;
      float tm = fmaxf(t0, t1);
#pragma unroll
      for (int off = 1; off < 16; off <<= 1)
        tm = fmaxf(tm, __shfl_xor(tm, off, 32));
      float newM = fmaxf(mrow[v], tm);
      float e0 = __expf(t0 - newM);
      float e1 = __expf(t1 - newM);
      float scale = __expf(mrow[v] - newM);
      mrow[v] = newM;
      float rs = e0 + e1;
#pragma unroll
      for (int off = 1; off < 16; off <<= 1)
        rs += __shfl_xor(rs, off, 32);
      lrow[v] = lrow[v] * scale + rs;
#pragma unroll
      for (int t = 0; t < 4; ++t) oacc[t][v] = oacc[t][v] * scale;
      int r = v + 8 * hf;
      pshare[wave][r][nl]      = (bf16)e0;
      pshare[wave][r][nl + 16] = (bf16)e1;
    }
    // P tile (16 queries x 32 keys) re-loaded in A-fragment layout
    union { v16bf v; __bf16 e[16]; } up;
#pragma unroll
    for (int i = 0; i < 8; ++i) {
      up.e[i]     = pshare[wave][nl][hf * 16 + i];
      up.e[8 + i] = pshare[wave][nl][hf * 16 + 8 + i];
    }
#pragma unroll
    for (int t = 0; t < 4; ++t) {
      const bf16* Vb = vT + ((size_t)(h * HD + t * 16)) * 2048 + (size_t)b * S + kb;
      v16bf bv = load_b_frag(Vb, 2048, lane);
      oacc[t] = wmma_bf16(up.v, bv, oacc[t]);
    }
  }
#pragma unroll
  for (int v = 0; v < 8; ++v) {
    float inv = 1.0f / lrow[v];
    size_t row = (size_t)b * S + qBase + v + 8 * hf;
#pragma unroll
    for (int t = 0; t < 4; ++t)
      obf[row * H + h * HD + t * 16 + nl] = (bf16)(oacc[t][v] * inv);
  }
}

// ---------------------------------------------------------------------------
// Launcher
// ---------------------------------------------------------------------------
extern "C" void kernel_launch(void* const* d_in, const int* in_sizes, int n_in,
                              void* d_out, int out_size, void* d_ws, size_t ws_size,
                              hipStream_t stream) {
  (void)in_sizes; (void)n_in; (void)out_size; (void)ws_size;
  const float* X       = (const float*)d_in[0];
  const float* i_w     = (const float*)d_in[1];
  const float* q_w     = (const float*)d_in[2];
  const float* k_w     = (const float*)d_in[3];
  const float* v_w     = (const float*)d_in[4];
  const float* out_w   = (const float*)d_in[5];
  const float* out_b   = (const float*)d_in[6];
  const float* ind_tab = (const float*)d_in[7];
  const float* key_tab = (const float*)d_in[8];
  const float* val_tab = (const float*)d_in[9];
  float* out = (float*)d_out;

  const int B = 2, S = 1024, Hd = 512, KK = 65536, M = B * S;

  char* ws = (char*)d_ws;
  size_t off = 0;
  auto take = [&](size_t bytes) -> char* {
    char* p = ws + off;
    off += (bytes + 255) & ~(size_t)255;
    return p;
  };
  bf16* Xbf      = (bf16*)take((size_t)M * Hd * 2);
  bf16* iwbf     = (bf16*)take((size_t)Hd * Hd * 2);
  bf16* qwbf     = (bf16*)take((size_t)Hd * Hd * 2);
  bf16* kwbf     = (bf16*)take((size_t)Hd * Hd * 2);
  bf16* vwbf     = (bf16*)take((size_t)Hd * Hd * 2);
  bf16* owbf     = (bf16*)take((size_t)Hd * Hd * 2);
  bf16* tabbf    = (bf16*)take((size_t)KK * Hd * 2);   // 64 MB, lives in L2
  bf16* idxbf    = (bf16*)take((size_t)M * Hd * 2);
  int*  choices  = (int*)take((size_t)M * 4);
  float* partVal = (float*)take((size_t)M * 8 * 4);
  int*  partIdx  = (int*)take((size_t)M * 8 * 4);
  bf16* ckbf     = (bf16*)take((size_t)M * Hd * 2);
  bf16* cvbf     = (bf16*)take((size_t)M * Hd * 2);
  bf16* gidxbf   = (bf16*)take((size_t)M * Hd * 2);
  bf16* chosenbf = (bf16*)take((size_t)M * Hd * 2);
  float* biasb   = (float*)take((size_t)B * S * S * 4);
  float* qf      = (float*)take((size_t)M * Hd * 4);
  float* kf      = (float*)take((size_t)M * Hd * 4);
  bf16* qb2      = (bf16*)take((size_t)M * Hd * 2);
  bf16* kb2      = (bf16*)take((size_t)M * Hd * 2);
  bf16* vTbf     = (bf16*)take((size_t)M * Hd * 2);
  bf16* obf      = (bf16*)take((size_t)M * Hd * 2);

  // 1. conversions to bf16
  cvt_bf16_kernel<<<2048, 256, 0, stream>>>(X, Xbf, (size_t)M * Hd);
  cvt_bf16_kernel<<<512, 256, 0, stream>>>(i_w, iwbf, (size_t)Hd * Hd);
  cvt_bf16_kernel<<<512, 256, 0, stream>>>(q_w, qwbf, (size_t)Hd * Hd);
  cvt_bf16_kernel<<<512, 256, 0, stream>>>(k_w, kwbf, (size_t)Hd * Hd);
  cvt_bf16_kernel<<<512, 256, 0, stream>>>(v_w, vwbf, (size_t)Hd * Hd);
  cvt_bf16_kernel<<<512, 256, 0, stream>>>(out_w, owbf, (size_t)Hd * Hd);
  cvt_bf16_kernel<<<4096, 256, 0, stream>>>(ind_tab, tabbf, (size_t)KK * Hd);

  const int gemmBlocks2048x512 = (M / 16) * (Hd / 16) / 8;   // 512
  const int gemmBlocks1024x1024 = (S / 16) * (S / 16) / 8;   // 512

  // 2. idx = sigmoid(X @ i_w^T)
  gemm_bf16_kernel<<<gemmBlocks2048x512, 256, 0, stream>>>(
      Xbf, iwbf, nullptr, idxbf, nullptr, M, Hd, Hd, 1, 0);

  // 3. fused sim GEMM + argmax (TDM-staged LDS panels), then slice reduce
  sim_argmax_tdm_kernel<<<128, 256, 0, stream>>>(idxbf, tabbf, partVal, partIdx);
  argmax_reduce_kernel<<<(M + 255) / 256, 256, 0, stream>>>(partVal, partIdx,
                                                            choices, M);

  // 4. gathers by choices
  gather_kernel<<<(M * Hd) / 256, 256, 0, stream>>>(
      choices, key_tab, val_tab, tabbf, ckbf, cvbf, gidxbf);

  // 5. cached codes ONLY for chosen rows: sigmoid(indices_tab[choices] @ i_w^T)
  gemm_bf16_kernel<<<gemmBlocks2048x512, 256, 0, stream>>>(
      gidxbf, iwbf, nullptr, chosenbf, nullptr, M, Hd, Hd, 1, 0);

  // 6. bias[b] = idx[b] @ chosen_idx[b]^T  (per batch)
  for (int b = 0; b < B; ++b)
    gemm_bf16_kernel<<<gemmBlocks1024x1024, 256, 0, stream>>>(
        idxbf + (size_t)b * S * Hd, chosenbf + (size_t)b * S * Hd,
        biasb + (size_t)b * S * S, nullptr, nullptr, S, S, Hd, 0, 0);

  // 7. projections
  gemm_bf16_kernel<<<gemmBlocks2048x512, 256, 0, stream>>>(
      Xbf, qwbf, qf, nullptr, nullptr, M, Hd, Hd, 0, 0);
  gemm_bf16_kernel<<<gemmBlocks2048x512, 256, 0, stream>>>(
      ckbf, kwbf, kf, nullptr, nullptr, M, Hd, Hd, 0, 0);
  gemm_bf16_kernel<<<gemmBlocks2048x512, 256, 0, stream>>>(
      cvbf, vwbf, nullptr, vTbf, nullptr, M, Hd, Hd, 0, 1);  // transposed store

  // 8. rotary on q/k, pack bf16
  rope_kernel<<<(M * Hd) / 256, 256, 0, stream>>>(qf, qb2, M * Hd);
  rope_kernel<<<(M * Hd) / 256, 256, 0, stream>>>(kf, kb2, M * Hd);

  // 9. attention (B*NH*S/16 = 1024 waves, 8 per block)
  attention_kernel<<<128, 256, 0, stream>>>(qb2, kb2, vTbf, biasb, obf);

  // 10. out = o @ out_w^T + out_b
  gemm_bf16_kernel<<<gemmBlocks2048x512, 256, 0, stream>>>(
      obf, owbf, out, nullptr, out_b, M, Hd, Hd, 0, 0);
}
